// Mamba2_71373766525029
// MI455X (gfx1250) — matrix operand order repaired and checked by reading, not compiled, not run on hardware
//
#include <hip/hip_runtime.h>

// ---------------------------------------------------------------------------
// Types for CDNA5 WMMA (wave32): v16bf = A/B fragment (16 bf16 = 8 VGPRs),
// v8f = C/D fragment (8 f32 = 8 VGPRs).
// ---------------------------------------------------------------------------
typedef __bf16 bf16_t;
typedef __bf16        v16bf __attribute__((ext_vector_type(16)));
typedef float         v8f   __attribute__((ext_vector_type(8)));
typedef unsigned int  v4u   __attribute__((ext_vector_type(4)));
typedef int           v4i   __attribute__((ext_vector_type(4)));

union Frag {
    v16bf v;
    v4u   u[2];   // two 16-byte chunks
};

// ---------------------------------------------------------------------------
// Problem constants (from the reference)
// ---------------------------------------------------------------------------
#define BATCH   4
#define SEQ_T   2048
#define DMODEL  2048
#define DINNER  4096
#define ROWS    (BATCH * SEQ_T)      // 8192 "M" rows for both GEMMs

// GEMM tiling: block tile 256x128, K-step 32; 8 waves of 64x64 each.
#define BM 256
#define BN 128
#define BK 32

// ---------------------------------------------------------------------------
// CDNA5 async global->LDS copy (ASYNCcnt path), with sync fallback.
// Probe round 2 revealed the builtin's signature: (AS1 int4*, AS3 int4*, Ii, Ii)
// ---------------------------------------------------------------------------
#if defined(__gfx1250__) && __has_builtin(__builtin_amdgcn_global_load_async_to_lds_b128)
#define USE_ASYNC_LDS 1
typedef __attribute__((address_space(1))) v4i* as1_v4i_ptr;
typedef __attribute__((address_space(3))) v4i* as3_v4i_ptr;
#else
#define USE_ASYNC_LDS 0
#endif

__device__ __forceinline__ void copy16_to_lds(const bf16_t* g, bf16_t* l)
{
#if USE_ASYNC_LDS
    __builtin_amdgcn_global_load_async_to_lds_b128(
        (as1_v4i_ptr)(g), (as3_v4i_ptr)(l), /*offset=*/0, /*cpol=*/0);
#else
    *(v4u*)l = *(const v4u*)g;
#endif
}

__device__ __forceinline__ void wait_async_copies()
{
#if USE_ASYNC_LDS
#if __has_builtin(__builtin_amdgcn_s_wait_asynccnt)
    __builtin_amdgcn_s_wait_asynccnt(0);
#else
    asm volatile("s_wait_asynccnt 0x0" ::: "memory");
#endif
#endif
}

// ---------------------------------------------------------------------------
// f32 -> bf16 conversion (grid-stride, RNE via clang __bf16 cast)
// ---------------------------------------------------------------------------
__global__ __launch_bounds__(256)
void k_f32_to_bf16(const float* __restrict__ s, bf16_t* __restrict__ d, long n)
{
    long i      = (long)blockIdx.x * blockDim.x + threadIdx.x;
    long stride = (long)gridDim.x * blockDim.x;
    for (; i < n; i += stride)
        d[i] = (bf16_t)s[i];
}

// ---------------------------------------------------------------------------
// C[M,N] = A[M,K] * B[N,K]^T   (A,B bf16 row-major, C f32)
// 256 threads = 8 waves. Block tile 256x128, K-step 32, double-buffered LDS
// fed by async global->LDS loads. Each wave owns a 64x64 sub-tile:
// 4 M-frags x 4 N-frags = 16 v_wmma_f32_16x16x32_bf16 per K-step.
// ---------------------------------------------------------------------------
__global__ __launch_bounds__(256, 1)
void k_wmma_gemm_bf16(const bf16_t* __restrict__ A,
                      const bf16_t* __restrict__ Bm,
                      float* __restrict__ C,
                      int M, int N, int K)
{
    __shared__ __align__(16) bf16_t lA[2][BM * BK];   // 2 x 16 KB
    __shared__ __align__(16) bf16_t lB[2][BN * BK];   // 2 x  8 KB

    const int tid  = threadIdx.x;
    const int wid  = tid >> 5;      // wave 0..7
    const int lane = tid & 31;
    const int lr   = lane & 15;
    const int half = lane >> 4;     // 0 or 1

    const int waveM = wid & 3;      // 4 waves along M (64 rows each)
    const int waveN = wid >> 2;     // 2 waves along N (64 cols each)

    const int tileM = blockIdx.y * BM;
    const int tileN = blockIdx.x * BN;

    const int nK = K / BK;

    // Tile copier: A tile = 1024 16B-chunks (4/thread), B tile = 512 (2/thread).
    auto issue_tile = [&](int kk, int buf) {
        const int k0 = kk * BK;
        #pragma unroll
        for (int i = 0; i < 4; ++i) {
            const int c    = tid + 256 * i;        // chunk id
            const int row  = c >> 2;               // 4 chunks per 32-elem row
            const int col8 = (c & 3) * 8;
            copy16_to_lds(A + (size_t)(tileM + row) * K + k0 + col8,
                          &lA[buf][row * BK + col8]);
        }
        #pragma unroll
        for (int i = 0; i < 2; ++i) {
            const int c    = tid + 256 * i;
            const int row  = c >> 2;
            const int col8 = (c & 3) * 8;
            copy16_to_lds(Bm + (size_t)(tileN + row) * K + k0 + col8,
                          &lB[buf][row * BK + col8]);
        }
        if (kk + 1 < nK) {   // L2 prefetch one more K-step ahead
            __builtin_prefetch(A  + (size_t)tileM * K + (size_t)(kk + 1) * BK, 0, 1);
            __builtin_prefetch(Bm + (size_t)tileN * K + (size_t)(kk + 1) * BK, 0, 1);
        }
    };

    v8f acc[4][4];
    #pragma unroll
    for (int i = 0; i < 4; ++i)
        #pragma unroll
        for (int j = 0; j < 4; ++j)
            #pragma unroll
            for (int e = 0; e < 8; ++e)
                acc[i][j][e] = 0.0f;

    issue_tile(0, 0);   // prologue

    for (int kk = 0; kk < nK; ++kk) {
        const int buf = kk & 1;

        wait_async_copies();     // this wave's tile-kk loads landed in LDS
        __syncthreads();         // whole tile kk visible; compute(kk-1) done

        if (kk + 1 < nK)
            issue_tile(kk + 1, buf ^ 1);   // overlap next loads with compute

        // Fragments per the CDNA5 16-bit VGPR layouts.
        Frag a[4], b[4];
        #pragma unroll
        for (int i = 0; i < 4; ++i) {
            const int row = waveM * 64 + i * 16 + lr;
            const int ka  = half * 8;        // A: K = ka..ka+7, ka+16..ka+23
            a[i].u[0] = *(const v4u*)&lA[buf][row * BK + ka];
            a[i].u[1] = *(const v4u*)&lA[buf][row * BK + ka + 16];
        }
        #pragma unroll
        for (int j = 0; j < 4; ++j) {
            const int col = waveN * 64 + j * 16 + lr;
            const int kb  = half * 8;        // B: K = kb..kb+7, kb+16..kb+23
            b[j].u[0] = *(const v4u*)&lB[buf][col * BK + kb];
            b[j].u[1] = *(const v4u*)&lB[buf][col * BK + kb + 16];
        }

        #pragma unroll
        for (int i = 0; i < 4; ++i)
            #pragma unroll
            for (int j = 0; j < 4; ++j)
                acc[i][j] = __builtin_amdgcn_wmma_f32_16x16x32_bf16(
                    /*neg_a=*/false, a[i].v, /*neg_b=*/false, b[j].v,
                    /*c_mod=*/(short)0, acc[i][j],
                    /*reuse_a=*/false, /*reuse_b=*/false);
    }

    // Epilogue: C/D layout -> VGPR v holds row (v + half*8), col = lr.
    #pragma unroll
    for (int i = 0; i < 4; ++i) {
        const int row0 = tileM + waveM * 64 + i * 16 + half * 8;
        #pragma unroll
        for (int j = 0; j < 4; ++j) {
            const int col = tileN + waveN * 64 + j * 16 + lr;
            #pragma unroll
            for (int v = 0; v < 8; ++v)
                C[(size_t)(row0 + v) * N + col] = acc[i][j][v];
        }
    }
}

// ---------------------------------------------------------------------------
// Depthwise causal conv (4 taps) + bias + silu(xc)*D*silu(gate), bf16 out.
// P is x_proj [ROWS, 2*DINNER]; H is [ROWS, DINNER] bf16.
// Thread mapping: d fastest -> all global accesses coalesced.
// ---------------------------------------------------------------------------
__global__ __launch_bounds__(256)
void k_conv_gate(const float* __restrict__ P,
                 const float* __restrict__ cw,
                 const float* __restrict__ cb,
                 const float* __restrict__ Dvec,
                 bf16_t* __restrict__ H)
{
    const int idx = blockIdx.x * 256 + threadIdx.x;   // 0 .. ROWS*DINNER-1
    const int d   = idx & (DINNER - 1);
    const int rt  = idx >> 12;                        // row in [0, ROWS)
    const int t   = rt & (SEQ_T - 1);

    float acc = cb[d];
    #pragma unroll
    for (int j = 0; j < 4; ++j) {
        const int tt = t - 3 + j;
        if (tt >= 0)
            acc += cw[d * 4 + j] * P[(size_t)(rt - 3 + j) * (2 * DINNER) + d];
    }
    const float g  = P[(size_t)rt * (2 * DINNER) + DINNER + d];
    const float sx = acc / (1.0f + __expf(-acc));
    const float sg = g   / (1.0f + __expf(-g));
    H[idx] = (bf16_t)(sx * Dvec[d] * sg);
}

// ---------------------------------------------------------------------------
// Host launcher
// ---------------------------------------------------------------------------
extern "C" void kernel_launch(void* const* d_in, const int* in_sizes, int n_in,
                              void* d_out, int out_size, void* d_ws, size_t ws_size,
                              hipStream_t stream)
{
    const float* x      = (const float*)d_in[0];   // [4,2048,2048]
    const float* w_in   = (const float*)d_in[1];   // [8192,2048]
    const float* w_out  = (const float*)d_in[2];   // [2048,4096]
    const float* conv_w = (const float*)d_in[3];   // [4096,1,4]
    const float* conv_b = (const float*)d_in[4];   // [4096]
    const float* Dvec   = (const float*)d_in[5];   // [4096]
    float*       out    = (float*)d_out;           // [4,2048,2048]

    const long N_X    = (long)ROWS * DMODEL;       // 16,777,216
    const long N_WIN  = (long)(2 * DINNER) * DMODEL;
    const long N_WOUT = (long)DMODEL * DINNER;
    const long N_P    = (long)ROWS * (2 * DINNER); // 67,108,864 f32
    const long N_H    = (long)ROWS * DINNER;       // 33,554,432 bf16

    char*  ws = (char*)d_ws;
    size_t o  = 0;
    auto take = [&](size_t bytes) { void* p = ws + o; o += (bytes + 255) & ~(size_t)255; return p; };

    bf16_t* xbf  = (bf16_t*)take((size_t)N_X * 2);
    bf16_t* wibf = (bf16_t*)take((size_t)N_WIN * 2);
    bf16_t* wobf = (bf16_t*)take((size_t)N_WOUT * 2);
    float*  P    = (float*) take((size_t)N_P * 4);
    bf16_t* H    = (bf16_t*)take((size_t)N_H * 2);
    (void)ws_size; (void)in_sizes; (void)n_in; (void)out_size;

    // 1) Convert operands to bf16.
    k_f32_to_bf16<<<8192, 256, 0, stream>>>(x,     xbf,  N_X);
    k_f32_to_bf16<<<8192, 256, 0, stream>>>(w_in,  wibf, N_WIN);
    k_f32_to_bf16<<<4096, 256, 0, stream>>>(w_out, wobf, N_WOUT);

    // 2) GEMM1: P[8192, 8192] = X[8192,2048] * W_in[8192,2048]^T
    k_wmma_gemm_bf16<<<dim3((2 * DINNER) / BN, ROWS / BM), 256, 0, stream>>>(
        xbf, wibf, P, ROWS, 2 * DINNER, DMODEL);

    // 3) conv + SiLU gating -> H (bf16)
    k_conv_gate<<<(int)(N_H / 256), 256, 0, stream>>>(P, conv_w, conv_b, Dvec, H);

    // 4) GEMM2: out[8192, 2048] = H[8192,4096] * W_out[2048,4096]^T
    k_wmma_gemm_bf16<<<dim3(DMODEL / BN, ROWS / BM), 256, 0, stream>>>(
        H, wobf, out, ROWS, DMODEL, DINNER);
}
